// Mamba_86260123174548
// MI455X (gfx1250) — matrix operand order repaired and checked by reading, not compiled, not run on hardware
//
#include <hip/hip_runtime.h>
#include <hip/hip_bf16.h>
#include <math.h>

// ---------- Problem constants (from reference) ----------
#define BATCH   2
#define SEQ     2048
#define D_MODEL 1024
#define D_STATE 16
#define D_CONV  4
#define D_INNER 2048            // 2*D_MODEL
#define DT_RANK 64              // D_MODEL/16
#define XPP     128             // padded x_proj rows (96 -> 128)
#define MROWS   (BATCH*SEQ)     // 4096

typedef __attribute__((ext_vector_type(8)))  float  v8f;
typedef __attribute__((ext_vector_type(16))) __bf16 v16bf;
typedef __attribute__((ext_vector_type(8)))  __bf16 v8bf;
typedef __attribute__((ext_vector_type(4)))  __bf16 v4bf;

#define DEVINL __device__ __forceinline__

// ---------------------------------------------------------------------
// CDNA5 async global->LDS copy (16B per lane), tracked by ASYNCcnt.
// VDST = wave-relative LDS byte offset, VADDR = 64-bit global address.
// ---------------------------------------------------------------------
DEVINL void async_cp16(unsigned lds_off, const void* gsrc) {
    asm volatile("global_load_async_to_lds_b128 %0, %1, off"
                 :: "v"(lds_off), "v"((unsigned long long)(uintptr_t)gsrc)
                 : "memory");
}
DEVINL void wait_async0() {
    asm volatile("s_wait_asynccnt 0x0" ::: "memory");
}
DEVINL unsigned lds_off32(const void* p) {   // shared aperture: offset in addr[31:0]
    return (unsigned)(uintptr_t)p;
}

// ---------------------------------------------------------------------
// Fragment builders per documented CDNA5 16-bit WMMA layouts.
// A 16x32: lanes 0-15 (row=r) hold K {0..7,16..23}; lanes 16-31 K {8..15,24..31}
// B 32x16: lanes 0-15 (col=r) hold K {0..7,16..23}; lanes 16-31 K {8..15,24..31}
// rowp points at element [row][0] of a 40-element-stride LDS row.
// ---------------------------------------------------------------------
DEVINL v16bf frag_a(const __bf16* rowp, int hl) {
    v8bf lo = *(const v8bf*)(rowp + hl * 8);
    v8bf hi = *(const v8bf*)(rowp + hl * 8 + 16);
    return __builtin_shufflevector(lo, hi, 0,1,2,3,4,5,6,7,8,9,10,11,12,13,14,15);
}
DEVINL v16bf frag_b(const __bf16* rowp, int hl) {
    v8bf lo = *(const v8bf*)(rowp + hl * 8);
    v8bf hi = *(const v8bf*)(rowp + hl * 8 + 16);
    return __builtin_shufflevector(lo, hi, 0,1,2,3,4,5,6,7,8,9,10,11,12,13,14,15);
}
DEVINL v8f wmma_bf16(v16bf a, v16bf b, v8f c) {
    return __builtin_amdgcn_wmma_f32_16x16x32_bf16(false, a, false, b, (short)0, c,
                                                   false, false);
}

// =====================================================================
// C[M,N] = A[M,K] * B[N,K]^T, all-bf16 operands, f32 accumulate.
// 256 threads = 8 waves; wave tile 32x32 (2x2 WMMA accumulators);
// block tile 128 (M) x 64 (N); KC=32; double-buffered LDS with async
// global->LDS staging (ASYNCcnt), one barrier per K-step.
// Requires M%128==0, N%64==0, K%32==0.
// =====================================================================
template <typename OUT_T>
__global__ __launch_bounds__(256) void gemm_bf16_wmma(
    const __bf16* __restrict__ A, int lda,
    const __bf16* __restrict__ B, int ldb,
    OUT_T* __restrict__ C, int ldc,
    int K)
{
    __shared__ __align__(16) __bf16 lsA[2][128][40];   // 40-elem stride keeps 16B align
    __shared__ __align__(16) __bf16 lsB[2][64][40];

    const int tid  = threadIdx.x;
    const int lane = tid & 31;
    const int wave = tid >> 5;
    const int wrow = wave >> 1;          // 0..3
    const int wcol = wave & 1;           // 0..1
    const int m0   = blockIdx.x * 128;
    const int n0   = blockIdx.y * 64;

    v8f acc00 = {}, acc01 = {}, acc10 = {}, acc11 = {};

    // staging map: one 16B chunk per (thread, slot)
    const int arow0 = tid >> 2;          // rows 0..63
    const int akg   = (tid & 3) * 8;     // 0,8,16,24
    const int hl    = lane >> 4;
    const int r     = lane & 15;

    const int nk = K >> 5;               // K/32 steps

    // per-thread source/destination bases
    const __bf16* srcA0 = &A[(size_t)(m0 + arow0) * lda + akg];
    const __bf16* srcA1 = &A[(size_t)(m0 + arow0 + 64) * lda + akg];
    const __bf16* srcB  = &B[(size_t)(n0 + arow0) * ldb + akg];

    // ---- preload K-step 0 into buffer 0 (async) ----
    async_cp16(lds_off32(&lsA[0][arow0     ][akg]), srcA0);
    async_cp16(lds_off32(&lsA[0][arow0 + 64][akg]), srcA1);
    async_cp16(lds_off32(&lsB[0][arow0     ][akg]), srcB);
    wait_async0();
    __syncthreads();

    for (int kk = 0; kk < nk; ++kk) {
        const int cur = kk & 1;

        // ---- async-stage next K-step into the other buffer ----
        if (kk + 1 < nk) {
            const int nxt = cur ^ 1;
            const size_t ka = (size_t)((kk + 1) << 5);
            async_cp16(lds_off32(&lsA[nxt][arow0     ][akg]), srcA0 + ka);
            async_cp16(lds_off32(&lsA[nxt][arow0 + 64][akg]), srcA1 + ka);
            async_cp16(lds_off32(&lsB[nxt][arow0     ][akg]), srcB  + ka);
        }
        if (kk + 2 < nk) {   // L2 hint for the K-step after next
            const size_t ka = (size_t)((kk + 2) << 5);
            __builtin_prefetch(srcA0 + ka, 0, 3);
            __builtin_prefetch(srcB  + ka, 0, 3);
        }

        // ---- 2x2 WMMA from current buffer ----
        v16bf a0 = frag_a(&lsA[cur][wrow * 32 + r     ][0], hl);
        v16bf a1 = frag_a(&lsA[cur][wrow * 32 + 16 + r][0], hl);
        v16bf b0 = frag_b(&lsB[cur][wcol * 32 + r     ][0], hl);
        v16bf b1 = frag_b(&lsB[cur][wcol * 32 + 16 + r][0], hl);

        acc00 = wmma_bf16(a0, b0, acc00);
        acc01 = wmma_bf16(a0, b1, acc01);
        acc10 = wmma_bf16(a1, b0, acc10);
        acc11 = wmma_bf16(a1, b1, acc11);

        // async copies must have landed before the barrier publishes the tile
        wait_async0();
        __syncthreads();
    }

    // ---- epilogue: VGPR v holds M=v (lanes 0-15) / M=v+8 (lanes 16-31), N=lane&15
#pragma unroll
    for (int i = 0; i < 2; ++i) {
#pragma unroll
        for (int j = 0; j < 2; ++j) {
            const v8f a = (i == 0) ? (j == 0 ? acc00 : acc01)
                                   : (j == 0 ? acc10 : acc11);
            const int col = n0 + wcol * 32 + j * 16 + r;
#pragma unroll
            for (int v = 0; v < 8; ++v) {
                const int row = m0 + wrow * 32 + i * 16 + v + hl * 8;
                C[(size_t)row * ldc + col] = (OUT_T)a[v];
            }
        }
    }
}

// =====================================================================
// f32 -> bf16 conversions (4 elems/thread, n % 1024 == 0)
// =====================================================================
__global__ __launch_bounds__(256) void cvt_f32_bf16(
    const float* __restrict__ src, __bf16* __restrict__ dst)
{
    const size_t e = ((size_t)blockIdx.x * 256 + threadIdx.x) * 4;
    const float4 f = *(const float4*)&src[e];
    v4bf o; o[0] = (__bf16)f.x; o[1] = (__bf16)f.y; o[2] = (__bf16)f.z; o[3] = (__bf16)f.w;
    *(v4bf*)&dst[e] = o;
}

// x_proj_w [96,2048] -> bf16 padded to [128,2048] (rows 96..127 = 0)
__global__ __launch_bounds__(256) void cvt_xprojw_pad(
    const float* __restrict__ src, __bf16* __restrict__ dst)
{
    const size_t e = ((size_t)blockIdx.x * 256 + threadIdx.x) * 4;
    const int row = (int)(e >> 11);
    v4bf o;
    if (row < 96) {
        const float4 f = *(const float4*)&src[e];
        o[0] = (__bf16)f.x; o[1] = (__bf16)f.y; o[2] = (__bf16)f.z; o[3] = (__bf16)f.w;
    } else {
        o[0] = (__bf16)0.f; o[1] = (__bf16)0.f; o[2] = (__bf16)0.f; o[3] = (__bf16)0.f;
    }
    *(v4bf*)&dst[e] = o;
}

// =====================================================================
// Causal depthwise conv1d (k=4) + SiLU, bf16 in (x view of xz), bf16 out
// =====================================================================
__global__ __launch_bounds__(256) void mamba_conv_silu(
    const __bf16* __restrict__ xz,      // [MROWS, 2*D_INNER]
    const float*  __restrict__ conv_w,  // [D_INNER, 4]
    const float*  __restrict__ conv_b,  // [D_INNER]
    __bf16* __restrict__ xc)            // [MROWS, D_INNER]
{
    const size_t gid = (size_t)blockIdx.x * blockDim.x + threadIdx.x;
    if (gid >= (size_t)MROWS * D_INNER) return;
    const int c = (int)(gid % D_INNER);
    const int m = (int)(gid / D_INNER);
    const int l = m % SEQ;

    float acc = conv_b[c];
#pragma unroll
    for (int j = 0; j < D_CONV; ++j) {
        const int ll = l - (D_CONV - 1) + j;
        if (ll >= 0) {
            acc += conv_w[c * D_CONV + j] *
                   (float)xz[(size_t)(m - (D_CONV - 1) + j) * (2 * D_INNER) + c];
        }
    }
    const float s = acc / (1.0f + __expf(-acc));      // SiLU
    xc[(size_t)m * D_INNER + c] = (__bf16)s;
}

// =====================================================================
// Fused selective scan: one lane per (batch, channel, state).
// 16-lane groups share one channel; h in a register; y reduced with
// 4 xor-shuffles. Fuses dt bias + softplus, x*D skip, SiLU(z) gate.
// =====================================================================
__global__ __launch_bounds__(256) void mamba_scan(
    const float*  __restrict__ dtraw,   // [MROWS, D_INNER] f32 (pre-bias)
    const float*  __restrict__ dt_b,    // [D_INNER]
    const float*  __restrict__ A_log,   // [D_INNER, D_STATE]
    const __bf16* __restrict__ xc,      // [MROWS, D_INNER] post conv+SiLU
    const __bf16* __restrict__ xp,      // [MROWS, XPP]: [64:80]=B, [80:96]=C
    const __bf16* __restrict__ xz,      // z view = xz[:, 2048:4096]
    const float*  __restrict__ Dvec,    // [D_INNER]
    __bf16* __restrict__ yg)            // [MROWS, D_INNER] gated scan out
{
    const int gid = blockIdx.x * blockDim.x + threadIdx.x;   // < 65536
    const int n   = gid & (D_STATE - 1);
    const int ch  = gid >> 4;            // 0..4095
    const int d   = ch & (D_INNER - 1);
    const int b   = ch >> 11;

    const float An   = -__expf(A_log[d * D_STATE + n]);
    const float Dd   = Dvec[d];
    const float bias = dt_b[d];

    float h = 0.0f;
    for (int l = 0; l < SEQ; ++l) {
        const size_t m = (size_t)b * SEQ + l;
        float dt = dtraw[m * D_INNER + d] + bias;            // broadcast in group
        dt = (dt > 20.0f) ? dt : log1pf(__expf(dt));         // softplus
        const float xv = (float)xc[m * D_INNER + d];
        const float Bv = (float)xp[m * XPP + DT_RANK + n];
        const float Cv = (float)xp[m * XPP + DT_RANK + D_STATE + n];

        const float dA = __expf(dt * An);
        h = dA * h + (dt * Bv) * xv;

        float p = h * Cv;
        p += __shfl_xor(p, 1, 32);
        p += __shfl_xor(p, 2, 32);
        p += __shfl_xor(p, 4, 32);
        p += __shfl_xor(p, 8, 32);                           // sum over 16 states

        if (n == 0) {
            const float zv = (float)xz[m * (2 * D_INNER) + D_INNER + d];
            const float yv = p + xv * Dd;
            yg[m * D_INNER + d] = (__bf16)(yv * (zv / (1.0f + __expf(-zv))));
        }
    }
}

// =====================================================================
// Launcher
// =====================================================================
extern "C" void kernel_launch(void* const* d_in, const int* in_sizes, int n_in,
                              void* d_out, int out_size, void* d_ws, size_t ws_size,
                              hipStream_t stream)
{
    (void)in_sizes; (void)n_in; (void)out_size; (void)ws_size;

    const float* hs        = (const float*)d_in[0];  // [2,2048,1024]
    const float* in_proj_w = (const float*)d_in[1];  // [4096,1024]
    const float* conv_w    = (const float*)d_in[2];  // [2048,1,4]
    const float* conv_b    = (const float*)d_in[3];  // [2048]
    const float* x_proj_w  = (const float*)d_in[4];  // [96,2048]
    const float* dt_proj_w = (const float*)d_in[5];  // [2048,64]
    const float* dt_proj_b = (const float*)d_in[6];  // [2048]
    const float* A_log     = (const float*)d_in[7];  // [2048,16]
    const float* Dv        = (const float*)d_in[8];  // [2048]
    const float* out_proj_w= (const float*)d_in[9];  // [1024,2048]
    float* out = (float*)d_out;                      // [2,2048,1024] f32

    // ---- workspace carve (256B aligned blocks) ----
    char* w = (char*)d_ws;
    auto carve = [&](size_t bytes) {
        char* p = w;
        w += (bytes + 255) & ~(size_t)255;
        return p;
    };
    __bf16* hs_bf   = (__bf16*)carve((size_t)MROWS * D_MODEL * 2);       // 8 MB
    __bf16* w_in_bf = (__bf16*)carve((size_t)2*D_INNER * D_MODEL * 2);   // 8 MB
    __bf16* w_xp_bf = (__bf16*)carve((size_t)XPP * D_INNER * 2);         // 0.5 MB
    __bf16* w_dt_bf = (__bf16*)carve((size_t)D_INNER * DT_RANK * 2);     // 0.25 MB
    __bf16* w_ou_bf = (__bf16*)carve((size_t)D_MODEL * D_INNER * 2);     // 4 MB
    __bf16* xz_bf   = (__bf16*)carve((size_t)MROWS * 2*D_INNER * 2);     // 32 MB
    __bf16* xc_bf   = (__bf16*)carve((size_t)MROWS * D_INNER * 2);       // 16 MB
    __bf16* xp_bf   = (__bf16*)carve((size_t)MROWS * XPP * 2);           // 1 MB
    float*  dtr     = (float*) carve((size_t)MROWS * D_INNER * 4);       // 32 MB
    __bf16* yg_bf   = (__bf16*)carve((size_t)MROWS * D_INNER * 2);       // 16 MB

    // 0) f32 -> bf16 conversions
    cvt_f32_bf16<<<(MROWS * D_MODEL) / 1024, 256, 0, stream>>>(hs, hs_bf);
    cvt_f32_bf16<<<(2 * D_INNER * D_MODEL) / 1024, 256, 0, stream>>>(in_proj_w, w_in_bf);
    cvt_xprojw_pad<<<(XPP * D_INNER) / 1024, 256, 0, stream>>>(x_proj_w, w_xp_bf);
    cvt_f32_bf16<<<(D_INNER * DT_RANK) / 1024, 256, 0, stream>>>(dt_proj_w, w_dt_bf);
    cvt_f32_bf16<<<(D_MODEL * D_INNER) / 1024, 256, 0, stream>>>(out_proj_w, w_ou_bf);

    // 1) xz = hs @ in_proj_w^T            [4096 x 4096], K=1024, bf16 out
    gemm_bf16_wmma<__bf16><<<dim3(MROWS / 128, (2 * D_INNER) / 64), 256, 0, stream>>>(
        hs_bf, D_MODEL, w_in_bf, D_MODEL, xz_bf, 2 * D_INNER, D_MODEL);

    // 2) xc = SiLU(causal_dwconv(x)),  x = xz[:, :2048]
    mamba_conv_silu<<<(MROWS * D_INNER) / 256, 256, 0, stream>>>(
        xz_bf, conv_w, conv_b, xc_bf);

    // 3) xp = xc @ x_proj_w_pad^T         [4096 x 128], K=2048, bf16 out
    gemm_bf16_wmma<__bf16><<<dim3(MROWS / 128, XPP / 64), 256, 0, stream>>>(
        xc_bf, D_INNER, w_xp_bf, D_INNER, xp_bf, XPP, D_INNER);

    // 4) dtr = xp[:, :64] @ dt_proj_w^T   [4096 x 2048], K=64, f32 out
    gemm_bf16_wmma<float><<<dim3(MROWS / 128, D_INNER / 64), 256, 0, stream>>>(
        xp_bf, XPP, w_dt_bf, DT_RANK, dtr, D_INNER, DT_RANK);

    // 5) fused selective scan + skip + gate -> yg (bf16)
    mamba_scan<<<(MROWS * D_STATE) / 256, 256, 0, stream>>>(
        dtr, dt_proj_b, A_log, xc_bf, xp_bf, xz_bf, Dv, yg_bf);

    // 6) out = yg @ out_proj_w^T          [4096 x 1024], K=2048, f32 out
    gemm_bf16_wmma<float><<<dim3(MROWS / 128, D_MODEL / 64), 256, 0, stream>>>(
        yg_bf, D_INNER, w_ou_bf, D_INNER, out, D_MODEL, D_INNER);
}
